// GatedGraphConvCNN_52321291600197
// MI455X (gfx1250) — matrix operand order repaired and checked by reading, not compiled, not run on hardware
//
#include <hip/hip_runtime.h>
#include <hip/hip_bf16.h>
#include <math.h>

// ---------------------------------------------------------------------------
// Types for CDNA5 WMMA (wave32): v_wmma_f32_16x16x32_bf16
// ---------------------------------------------------------------------------
typedef __attribute__((ext_vector_type(16))) __bf16          v16bf;
typedef __attribute__((ext_vector_type(8)))  float           v8f;
typedef __attribute__((ext_vector_type(8)))  unsigned short  ushort8;
typedef __attribute__((ext_vector_type(4)))  unsigned short  ushort4v;

union Frag { v16bf v; ushort8 h[2]; };

#define HDIM 128
#define GDIM 16
#define ELEC 32
#define BSZ  128
#define LDIM 63          // (128-3)/2 + 1
#define NCLS 3

// ---------------------------------------------------------------------------
// bf16 helpers (RNE convert / widen)
// ---------------------------------------------------------------------------
__device__ __forceinline__ unsigned short f2bf(float f) {
  unsigned int u = __float_as_uint(f);
  unsigned int r = u + 0x7FFFu + ((u >> 16) & 1u);
  return (unsigned short)(r >> 16);
}
__device__ __forceinline__ float bf2f(unsigned short h) {
  return __uint_as_float(((unsigned int)h) << 16);
}
__device__ __forceinline__ float sigmoidf(float x) {
  return 1.0f / (1.0f + __expf(-x));
}

// ---------------------------------------------------------------------------
// WMMA fragment loaders.
//   A tile (16x32 bf16, ISA layout): lane l -> row (l&15); elements 0..7 hold
//   K = kb+off..kb+off+7, elements 8..15 hold K = kb+off+16.., off=(l>=16)*8.
//   B pack: pre-swizzled so each lane's 16 elements are 32 contiguous bytes.
// ---------------------------------------------------------------------------
__device__ __forceinline__ void load_a4(Frag a[4], const unsigned short* __restrict__ rowp, int off) {
#pragma unroll
  for (int kt = 0; kt < 4; ++kt) {
    a[kt].h[0] = *(const ushort8*)(rowp + kt * 32 + off);
    a[kt].h[1] = *(const ushort8*)(rowp + kt * 32 + off + 16);
  }
}

__device__ __forceinline__ v8f wmma_tile(const Frag a[4], const unsigned short* __restrict__ pack,
                                         int ntile, int lane) {
  v8f acc = {0.f, 0.f, 0.f, 0.f, 0.f, 0.f, 0.f, 0.f};
  const ushort8* bp = (const ushort8*)pack + (size_t)ntile * 256 + lane * 2; // ntile*2048 ushorts
#pragma unroll
  for (int kt = 0; kt < 4; ++kt) {
    Frag b;
    b.h[0] = bp[kt * 64 + 0];
    b.h[1] = bp[kt * 64 + 1];
    acc = __builtin_amdgcn_wmma_f32_16x16x32_bf16(false, a[kt].v, false, b.v,
                                                  (short)0, acc, false, false);
  }
  return acc;
}

// ---------------------------------------------------------------------------
// Weight pre-swizzle into WMMA B layout.
//   pack index = ((ntile*4 + kt)*32 + lane)*16 + e
//   logical K  = kt*32 + (e&7) + 8*(lane>=16) + 16*(e>>3), logical N = ntile*16 + (lane&15)
// ---------------------------------------------------------------------------
__global__ void pack_b_direct(const float* __restrict__ W, unsigned short* __restrict__ pack,
                              int ntiles) {          // B[k][n] = W[k*Ncols + n], K=128
  int idx = blockIdx.x * 256 + threadIdx.x;
  int total = ntiles * 2048;
  if (idx >= total) return;
  int e = idx & 15, lane = (idx >> 4) & 31, kt = (idx >> 9) & 3, nt = idx >> 11;
  int K = kt * 32 + (e & 7) + ((lane >= 16) ? 8 : 0) + ((e >> 3) << 4);
  int Nn = nt * 16 + (lane & 15);
  pack[idx] = f2bf(W[(size_t)K * (ntiles * 16) + Nn]);
}

__global__ void pack_b_transpose(const float* __restrict__ W, unsigned short* __restrict__ pack,
                                 int ntiles) {       // B[k][n] = W[n*128 + k]
  int idx = blockIdx.x * 256 + threadIdx.x;
  int total = ntiles * 2048;
  if (idx >= total) return;
  int e = idx & 15, lane = (idx >> 4) & 31, kt = (idx >> 9) & 3, nt = idx >> 11;
  int K = kt * 32 + (e & 7) + ((lane >= 16) ? 8 : 0) + ((e >> 3) << 4);
  int Nn = nt * 16 + (lane & 15);
  pack[idx] = f2bf(W[(size_t)Nn * HDIM + K]);
}

// ---------------------------------------------------------------------------
// h init: pad x[N,5] -> h[N,128]
// ---------------------------------------------------------------------------
__global__ void pad_kernel(const float* __restrict__ x, float* __restrict__ h, int N) {
  int i = blockIdx.x * 256 + threadIdx.x;
  if (i >= N * HDIM) return;
  int f = i & (HDIM - 1), n = i >> 7;
  h[i] = (f < 5) ? x[(size_t)n * 5 + f] : 0.0f;
}

__global__ void cvt_bf16_kernel(const float* __restrict__ in, unsigned short* __restrict__ out, int n) {
  int i = blockIdx.x * 256 + threadIdx.x;
  if (i < n) out[i] = f2bf(in[i]);
}

__global__ void zero_f32_kernel(float* __restrict__ p, int n) {
  int i = blockIdx.x * 256 + threadIdx.x;
  if (i < n) p[i] = 0.0f;
}

// ---------------------------------------------------------------------------
// m = h @ W (128x128), output directly in bf16 (halves edge-gather traffic).
// One block = 16 nodes, 8 waves = 8 column tiles of 16.
// ---------------------------------------------------------------------------
__global__ void __launch_bounds__(256) gemm_m_kernel(const unsigned short* __restrict__ hb,
                                                     const unsigned short* __restrict__ pack,
                                                     unsigned short* __restrict__ mb) {
  const int lane = threadIdx.x & 31;
  const int wave = threadIdx.x >> 5;     // N tile
  const int nb   = blockIdx.x << 4;
  const int row  = nb + (lane & 15);
  const int off  = (lane >= 16) ? 8 : 0;
  Frag a[4];
  load_a4(a, hb + (size_t)row * HDIM, off);
  v8f acc = wmma_tile(a, pack, wave, lane);
  const int hi  = (lane >= 16) ? 8 : 0;
  const int col = (wave << 4) + (lane & 15);
#pragma unroll
  for (int e = 0; e < 8; ++e)
    mb[(size_t)(nb + e + hi) * HDIM + col] = f2bf(acc[e]);
}

// ---------------------------------------------------------------------------
// Edge scatter: agg[dst] += bf2f(m[src]) * w.  One wave per edge, 4 feats/lane.
// agg (32MB) and mb (16MB) are L2-resident on MI455X (192MB L2).
// ---------------------------------------------------------------------------
__global__ void __launch_bounds__(256) edge_scatter_kernel(const int* __restrict__ ei,
                                                           const float* __restrict__ ea,
                                                           const unsigned short* __restrict__ mb,
                                                           float* __restrict__ agg, int E) {
  unsigned int gid = blockIdx.x * 256 + threadIdx.x;
  int eid = gid >> 5;
  int lane = gid & 31;
  if (eid >= E) return;
  int src = ei[eid];
  int dst = ei[(size_t)E + eid];
  float w = ea[eid];
  ushort4v m4 = *(const ushort4v*)(mb + (size_t)src * HDIM + lane * 4);
  float* ap = agg + (size_t)dst * HDIM + lane * 4;
  atomicAdd(ap + 0, bf2f(m4[0]) * w);
  atomicAdd(ap + 1, bf2f(m4[1]) * w);
  atomicAdd(ap + 2, bf2f(m4[2]) * w);
  atomicAdd(ap + 3, bf2f(m4[3]) * w);
}

// ---------------------------------------------------------------------------
// Fused GRU: per block 16 nodes; wave w owns cols [16w,16w+16).
// Computes gi_r/z/n = agg@Wih.T and gh_r/z/n = h@Whh.T as six 16x16 WMMA
// tiles (K=128) and applies the full GRU update in registers -> no 192MB of
// gi/gh intermediates touching memory.
// Wih/Whh packs have 24 N-tiles: [0..7]=r, [8..15]=z, [16..23]=n slices.
// ---------------------------------------------------------------------------
__global__ void __launch_bounds__(256) gru_fused_kernel(const unsigned short* __restrict__ ab,
                                                        const unsigned short* __restrict__ hb,
                                                        const unsigned short* __restrict__ packIh,
                                                        const unsigned short* __restrict__ packHh,
                                                        const float* __restrict__ bih,
                                                        const float* __restrict__ bhh,
                                                        const float* __restrict__ hcur,
                                                        float* __restrict__ hnext) {
  const int lane = threadIdx.x & 31;
  const int wave = threadIdx.x >> 5;
  const int nb   = blockIdx.x << 4;
  const int row  = nb + (lane & 15);
  const int off  = (lane >= 16) ? 8 : 0;

  Frag aA[4], aH[4];
  load_a4(aA, ab + (size_t)row * HDIM, off);   // A for gi  (agg)
  load_a4(aH, hb + (size_t)row * HDIM, off);   // A for gh  (h)

  v8f ir = wmma_tile(aA, packIh, wave,      lane);
  v8f iz = wmma_tile(aA, packIh, 8 + wave,  lane);
  v8f in_ = wmma_tile(aA, packIh, 16 + wave, lane);
  v8f hr = wmma_tile(aH, packHh, wave,      lane);
  v8f hz = wmma_tile(aH, packHh, 8 + wave,  lane);
  v8f hn = wmma_tile(aH, packHh, 16 + wave, lane);

  const int col = (wave << 4) + (lane & 15);
  const float bir = bih[col],            bhr = bhh[col];
  const float biz = bih[HDIM + col],     bhz = bhh[HDIM + col];
  const float bin_ = bih[2 * HDIM + col], bhn = bhh[2 * HDIM + col];
  const int hi = (lane >= 16) ? 8 : 0;

#pragma unroll
  for (int e = 0; e < 8; ++e) {
    int node = nb + e + hi;
    float r  = sigmoidf(ir[e] + bir + hr[e] + bhr);
    float z  = sigmoidf(iz[e] + biz + hz[e] + bhz);
    float nn = tanhf(in_[e] + bin_ + r * (hn[e] + bhn));
    float ho = hcur[(size_t)node * HDIM + col];
    hnext[(size_t)node * HDIM + col] = (1.0f - z) * nn + z * ho;
  }
}

// ---------------------------------------------------------------------------
// relu + reshape + conv1d(G->1, k=3, stride 2) + relu.  Output c[b][e*63+l].
// ---------------------------------------------------------------------------
__global__ void conv_kernel(const float* __restrict__ h, const float* __restrict__ cw,
                            const float* __restrict__ cb, float* __restrict__ c) {
  int idx = blockIdx.x * 256 + threadIdx.x;
  if (idx >= BSZ * ELEC * LDIM) return;
  int l = idx % LDIM;
  int n = idx / LDIM;
  int e = n & (ELEC - 1);
  int b = n >> 5;
  float s = cb[0];
#pragma unroll
  for (int g = 0; g < GDIM; ++g) {
    const float* hp = h + ((size_t)((b * GDIM + g) * ELEC + e)) * HDIM + 2 * l;
    s += fmaxf(hp[0], 0.f) * cw[g * 3 + 0]
       + fmaxf(hp[1], 0.f) * cw[g * 3 + 1]
       + fmaxf(hp[2], 0.f) * cw[g * 3 + 2];
  }
  c[(size_t)b * (ELEC * LDIM) + e * LDIM + l] = fmaxf(s, 0.f);
}

// ---------------------------------------------------------------------------
// lin1: o1[b][j] = relu(c[b,:] . w1[j,:] + b1[j]).  One wave per (b,j).
// ---------------------------------------------------------------------------
__global__ void lin1_kernel(const float* __restrict__ c, const float* __restrict__ w1,
                            const float* __restrict__ b1, float* __restrict__ o1) {
  unsigned int gid = blockIdx.x * 256 + threadIdx.x;
  int wid = gid >> 5;
  int lane = gid & 31;
  if (wid >= BSZ * HDIM) return;
  int b = wid >> 7;
  int j = wid & (HDIM - 1);
  const int F = ELEC * LDIM;  // 2016
  float s = 0.f;
  for (int t = lane; t < F; t += 32)
    s += c[(size_t)b * F + t] * w1[(size_t)j * F + t];
#pragma unroll
  for (int d = 16; d > 0; d >>= 1) s += __shfl_down(s, d, 32);
  if (lane == 0) o1[b * HDIM + j] = fmaxf(s + b1[j], 0.f);
}

// ---------------------------------------------------------------------------
// lin2 + softmax (tiny).  One thread per batch row.
// ---------------------------------------------------------------------------
__global__ void lin2_softmax_kernel(const float* __restrict__ o1, const float* __restrict__ w2,
                                    const float* __restrict__ b2, float* __restrict__ out) {
  int b = threadIdx.x;
  if (b >= BSZ) return;
  float v[NCLS];
  float m = -1e30f;
#pragma unroll
  for (int cc = 0; cc < NCLS; ++cc) {
    float s = b2[cc];
    for (int k = 0; k < HDIM; ++k) s += o1[b * HDIM + k] * w2[cc * HDIM + k];
    v[cc] = s;
    m = fmaxf(m, s);
  }
  float sum = 0.f;
#pragma unroll
  for (int cc = 0; cc < NCLS; ++cc) { v[cc] = __expf(v[cc] - m); sum += v[cc]; }
#pragma unroll
  for (int cc = 0; cc < NCLS; ++cc) out[b * NCLS + cc] = v[cc] / sum;
}

// ---------------------------------------------------------------------------
// Host driver
// ---------------------------------------------------------------------------
extern "C" void kernel_launch(void* const* d_in, const int* in_sizes, int n_in,
                              void* d_out, int out_size, void* d_ws, size_t ws_size,
                              hipStream_t stream) {
  (void)n_in; (void)out_size; (void)ws_size;
  const float* x   = (const float*)d_in[0];
  const int*   ei  = (const int*)  d_in[1];
  const float* ea  = (const float*)d_in[2];
  const float* gw  = (const float*)d_in[4];   // (2,128,128)
  const float* wih = (const float*)d_in[5];   // (384,128)
  const float* whh = (const float*)d_in[6];   // (384,128)
  const float* bih = (const float*)d_in[7];
  const float* bhh = (const float*)d_in[8];
  const float* cw  = (const float*)d_in[9];
  const float* cb  = (const float*)d_in[10];
  const float* w1  = (const float*)d_in[11];
  const float* b1  = (const float*)d_in[12];
  const float* w2  = (const float*)d_in[13];
  const float* b2  = (const float*)d_in[14];

  const int N = in_sizes[0] / 5;     // 65536
  const int E = in_sizes[1] / 2;     // 1048576

  // workspace carve (256B aligned)
  char* ws = (char*)d_ws;
  size_t off = 0;
  auto carve = [&](size_t bytes) -> char* {
    char* p = ws + off;
    off += (bytes + 255) & ~(size_t)255;
    return p;
  };
  float*          hA     = (float*)carve((size_t)N * HDIM * 4);
  float*          hB     = (float*)carve((size_t)N * HDIM * 4);
  float*          agg    = (float*)carve((size_t)N * HDIM * 4);
  unsigned short* hb16   = (unsigned short*)carve((size_t)N * HDIM * 2);
  unsigned short* sb16   = (unsigned short*)carve((size_t)N * HDIM * 2); // mb, then aggb
  unsigned short* packG0 = (unsigned short*)carve(8  * 2048 * 2);
  unsigned short* packG1 = (unsigned short*)carve(8  * 2048 * 2);
  unsigned short* packIh = (unsigned short*)carve(24 * 2048 * 2);
  unsigned short* packHh = (unsigned short*)carve(24 * 2048 * 2);
  float*          convc  = (float*)carve((size_t)BSZ * ELEC * LDIM * 4);
  float*          o1     = (float*)carve((size_t)BSZ * HDIM * 4);

  const int nh = N * HDIM;
  const int bEl = (nh + 255) / 256;

  // weight pre-swizzle into WMMA B layout (cheap, done every call: deterministic)
  pack_b_direct<<<(8 * 2048 + 255) / 256, 256, 0, stream>>>(gw,                 packG0, 8);
  pack_b_direct<<<(8 * 2048 + 255) / 256, 256, 0, stream>>>(gw + HDIM * HDIM,   packG1, 8);
  pack_b_transpose<<<(24 * 2048 + 255) / 256, 256, 0, stream>>>(wih, packIh, 24);
  pack_b_transpose<<<(24 * 2048 + 255) / 256, 256, 0, stream>>>(whh, packHh, 24);

  // h0 = pad(x)
  pad_kernel<<<bEl, 256, 0, stream>>>(x, hA, N);

  float* hcur = hA;
  float* hnext = hB;
  const unsigned short* packs[2] = {packG0, packG1};

  for (int layer = 0; layer < 2; ++layer) {
    cvt_bf16_kernel<<<bEl, 256, 0, stream>>>(hcur, hb16, nh);
    gemm_m_kernel<<<N / 16, 256, 0, stream>>>(hb16, packs[layer], sb16);       // m (bf16)
    zero_f32_kernel<<<bEl, 256, 0, stream>>>(agg, nh);
    edge_scatter_kernel<<<(E * 32 + 255) / 256, 256, 0, stream>>>(ei, ea, sb16, agg, E);
    cvt_bf16_kernel<<<bEl, 256, 0, stream>>>(agg, sb16, nh);                   // aggb reuses sb16
    gru_fused_kernel<<<N / 16, 256, 0, stream>>>(sb16, hb16, packIh, packHh,
                                                 bih, bhh, hcur, hnext);
    float* t = hcur; hcur = hnext; hnext = t;
  }

  // head: relu -> conv1d -> relu -> lin1 -> relu -> lin2 -> softmax
  conv_kernel<<<(BSZ * ELEC * LDIM + 255) / 256, 256, 0, stream>>>(hcur, cw, cb, convc);
  lin1_kernel<<<(BSZ * HDIM * 32 + 255) / 256, 256, 0, stream>>>(convc, w1, b1, o1);
  lin2_softmax_kernel<<<1, 128, 0, stream>>>(o1, w2, b2, (float*)d_out);
}
